// GatedGIN_di_52338471469200
// MI455X (gfx1250) — matrix operand order repaired and checked by reading, not compiled, not run on hardware
//
#include <hip/hip_runtime.h>
#include <hip/hip_bf16.h>
#include <math.h>

typedef __bf16 bf16_t;
typedef __attribute__((ext_vector_type(16))) bf16_t v16bf;
typedef __attribute__((ext_vector_type(8)))  bf16_t v8bf;
typedef __attribute__((ext_vector_type(8)))  float  v8f;

// ---------------------------------------------------------------------------
// WMMA tile loaders (contiguous k order, same for A and B, wave32)
// A-matrix 16x32 bf16: lane l -> row m=l&15; halves j = K (k0+(l>>4)*16+j)
// ---------------------------------------------------------------------------
__device__ __forceinline__ v16bf load_a_tile(const float* __restrict__ A, int ldA,
                                             int row0, int k0, int lane) {
  const int m  = lane & 15;
  const int hi = lane >> 4;
  const float* ap = A + (size_t)(row0 + m) * ldA + k0;
  v16bf a;
#pragma unroll
  for (int j = 0; j < 8; ++j) a[j] = (bf16_t)ap[hi * 16 + j];
#pragma unroll
  for (int j = 0; j < 8; ++j) a[8 + j] = (bf16_t)ap[hi * 16 + 8 + j];
  return a;
}

__device__ __forceinline__ v16bf load_a_tile(const bf16_t* __restrict__ A, int ldA,
                                             int row0, int k0, int lane) {
  const int m  = lane & 15;
  const int hi = lane >> 4;
  const bf16_t* ap = A + (size_t)(row0 + m) * ldA + k0;
  const v8bf lo = *(const v8bf*)(ap + hi * 16);       // 16B aligned
  const v8bf up = *(const v8bf*)(ap + hi * 16 + 8);   // 16B aligned
  v16bf a;
#pragma unroll
  for (int j = 0; j < 8; ++j) { a[j] = lo[j]; a[8 + j] = up[j]; }
  return a;
}

// B-matrix 32x16 bf16 (= W^T tile, W row-major [OUT,K] pre-converted to bf16):
// lane l holds column n=l&15 of B = row (n0+n) of W; halves j -> K = k0+(l>>4)*16+j
__device__ __forceinline__ v16bf load_b_tile(const bf16_t* __restrict__ W, int ldW,
                                             int n0, int k0, int lane) {
  const int n  = lane & 15;
  const int hi = lane >> 4;
  return *(const v16bf*)(W + (size_t)(n0 + n) * ldW + k0 + hi * 16);  // 32B aligned
}

__device__ __forceinline__ v8f wmma_bf16(v16bf a, v16bf b, v8f c) {
  return __builtin_amdgcn_wmma_f32_16x16x32_bf16(false, a, false, b, (short)0, c,
                                                 false, false);
}

// ---------------------------------------------------------------------------
// C[N,OUT] = A[N,K] @ W[OUT,K]^T + bias.  A is fp32 or bf16 (template),
// W pre-converted bf16.  Output: fp32 (Cf) and/or bf16 (Cb), either may be null.
// One wave -> 16 rows x OUT cols; 4 waves / block.
// ---------------------------------------------------------------------------
template <int K, int OUT, typename AT>
__global__ __launch_bounds__(128) void gemm_bias_kernel(
    const AT* __restrict__ A, const bf16_t* __restrict__ Wb,
    const float* __restrict__ bias, float* __restrict__ Cf,
    bf16_t* __restrict__ Cb, int nrows) {
  constexpr int NT = OUT / 16;
  const int lane = threadIdx.x & 31;
  const int wave = threadIdx.x >> 5;
  const int row0 = (blockIdx.x * 4 + wave) * 16;
  if (row0 >= nrows) return;

  v8f acc[NT];
#pragma unroll
  for (int t = 0; t < NT; ++t) {
    const float bv = bias[t * 16 + (lane & 15)];
#pragma unroll
    for (int r = 0; r < 8; ++r) acc[t][r] = bv;
  }

#pragma unroll
  for (int k0 = 0; k0 < K; k0 += 32) {
    const v16bf a = load_a_tile(A, K, row0, k0, lane);
#pragma unroll
    for (int t = 0; t < NT; ++t) {
      const v16bf b = load_b_tile(Wb, K, t * 16, k0, lane);
      acc[t] = wmma_bf16(a, b, acc[t]);
    }
  }

  const int col = lane & 15;
  const int hi  = lane >> 4;
  if (Cf) {
#pragma unroll
    for (int t = 0; t < NT; ++t)
#pragma unroll
      for (int r = 0; r < 8; ++r)
        Cf[(size_t)(row0 + r + 8 * hi) * OUT + t * 16 + col] = acc[t][r];
  }
  if (Cb) {
#pragma unroll
    for (int t = 0; t < NT; ++t)
#pragma unroll
      for (int r = 0; r < 8; ++r)
        Cb[(size_t)(row0 + r + 8 * hi) * OUT + t * 16 + col] = (bf16_t)acc[t][r];
  }
}

// ---------------------------------------------------------------------------
// Fused GRU: gi = X@W_ih^T + b_ih ; gh = H@W_hh^T + b_hh (24 WMMA acc tiles in
// registers), then gating in-register.  X/H A-tiles bf16, h-state blend fp32.
// ---------------------------------------------------------------------------
__device__ __forceinline__ float sigm(float x) { return 1.0f / (1.0f + expf(-x)); }

__global__ __launch_bounds__(64) void gru_kernel(
    const bf16_t* __restrict__ Xb, const bf16_t* __restrict__ Hb,
    const float* __restrict__ Hf, const bf16_t* __restrict__ Wihb,
    const bf16_t* __restrict__ Whhb, const float* __restrict__ bih,
    const float* __restrict__ bhh, bf16_t* __restrict__ Outb, int nrows) {
  const int lane = threadIdx.x & 31;
  const int wave = threadIdx.x >> 5;
  const int row0 = (blockIdx.x * 2 + wave) * 16;
  if (row0 >= nrows) return;

  v8f gi[12], gh[12];
#pragma unroll
  for (int t = 0; t < 12; ++t) {
    const float bi = bih[t * 16 + (lane & 15)];
    const float bh = bhh[t * 16 + (lane & 15)];
#pragma unroll
    for (int r = 0; r < 8; ++r) { gi[t][r] = bi; gh[t][r] = bh; }
  }

#pragma unroll
  for (int k0 = 0; k0 < 64; k0 += 32) {
    const v16bf ax = load_a_tile(Xb, 64, row0, k0, lane);
    const v16bf ah = load_a_tile(Hb, 64, row0, k0, lane);
#pragma unroll
    for (int t = 0; t < 12; ++t) {
      const v16bf bi = load_b_tile(Wihb, 64, t * 16, k0, lane);
      gi[t] = wmma_bf16(ax, bi, gi[t]);
      const v16bf bh = load_b_tile(Whhb, 64, t * 16, k0, lane);
      gh[t] = wmma_bf16(ah, bh, gh[t]);
    }
  }

  const int col = lane & 15;
  const int hi  = lane >> 4;
#pragma unroll
  for (int t = 0; t < 4; ++t) {
#pragma unroll
    for (int r = 0; r < 8; ++r) {
      const float rg = sigm(gi[t][r]     + gh[t][r]);       // reset gate
      const float zg = sigm(gi[t + 4][r] + gh[t + 4][r]);   // update gate
      const float ng = tanhf(gi[t + 8][r] + rg * gh[t + 8][r]);
      const size_t idx = (size_t)(row0 + r + 8 * hi) * 64 + t * 16 + col;
      const float hv = Hf[idx];
      Outb[idx] = (bf16_t)((1.0f - zg) * ng + zg * hv);
    }
  }
}

// ---------------------------------------------------------------------------
// Edge propagation: 16 lanes per edge, float4 gather, scatter-add with
// global_atomic_add_f32 into [N,128] accumulator (col half selected by colOff)
// ---------------------------------------------------------------------------
__global__ __launch_bounds__(256) void prop_kernel(
    const float* __restrict__ x, const int* __restrict__ eidx,
    const float* __restrict__ ew, float* __restrict__ acc, int colOff, int E) {
  const long long gid = (long long)blockIdx.x * blockDim.x + threadIdx.x;
  const long long e = gid >> 4;
  const int q = (int)(gid & 15);
  if (e >= E) return;
  const int s = eidx[e];
  const int d = eidx[(long long)E + e];
  const float w = ew[e];
  const float4 v = *(const float4*)(x + (size_t)s * 64 + q * 4);
  float* dst = acc + (size_t)d * 128 + colOff + q * 4;
  atomicAdd(dst + 0, v.x * w);
  atomicAdd(dst + 1, v.y * w);
  atomicAdd(dst + 2, v.z * w);
  atomicAdd(dst + 3, v.w * w);
}

__global__ __launch_bounds__(256) void zero4_kernel(float4* __restrict__ p, size_t n4) {
  const size_t i = (size_t)blockIdx.x * blockDim.x + threadIdx.x;
  if (i < n4) p[i] = make_float4(0.f, 0.f, 0.f, 0.f);
}

__global__ __launch_bounds__(256) void f32_to_bf16_kernel(const float* __restrict__ src,
                                                          bf16_t* __restrict__ dst, int n) {
  const int i = blockIdx.x * blockDim.x + threadIdx.x;
  if (i < n) dst[i] = (bf16_t)src[i];
}

__global__ __launch_bounds__(256) void logsoftmax16_kernel(float* __restrict__ y, int n) {
  const int i = blockIdx.x * blockDim.x + threadIdx.x;
  if (i >= n) return;
  float* p = y + (size_t)i * 16;
  float v[16];
  float m = -INFINITY;
#pragma unroll
  for (int j = 0; j < 16; ++j) { v[j] = p[j]; m = fmaxf(m, v[j]); }
  float s = 0.f;
#pragma unroll
  for (int j = 0; j < 16; ++j) s += expf(v[j] - m);
  const float ls = logf(s) + m;
#pragma unroll
  for (int j = 0; j < 16; ++j) p[j] = v[j] - ls;
}

// ---------------------------------------------------------------------------
extern "C" void kernel_launch(void* const* d_in, const int* in_sizes, int n_in,
                              void* d_out, int out_size, void* d_ws, size_t ws_size,
                              hipStream_t stream) {
  (void)n_in; (void)out_size; (void)ws_size;
  const float* x_in    = (const float*)d_in[0];
  const int*   ei      = (const int*)  d_in[1];
  const float* ew      = (const float*)d_in[2];
  const int*   ei_r    = (const int*)  d_in[3];
  const float* ew_r    = (const float*)d_in[4];
  const float* W_first = (const float*)d_in[5];
  const float* b_first = (const float*)d_in[6];
  const float* W_con1  = (const float*)d_in[7];
  const float* b_con1  = (const float*)d_in[8];
  const float* W_con2  = (const float*)d_in[9];
  const float* b_con2  = (const float*)d_in[10];
  const float* W_lin1  = (const float*)d_in[11];
  const float* b_lin1  = (const float*)d_in[12];
  const float* W_out   = (const float*)d_in[13];
  const float* b_out   = (const float*)d_in[14];
  const float* W_ih    = (const float*)d_in[15];
  const float* W_hh    = (const float*)d_in[16];
  const float* b_ih    = (const float*)d_in[17];
  const float* b_hh    = (const float*)d_in[18];

  const int F = 128, H = 64;
  const int N = in_sizes[0] / F;
  const int E = in_sizes[1] / 2;
  const size_t NH = (size_t)N * H;

  // fp32 buffers (prop gather/scatter + GRU state blend need full precision)
  float* buf_h = (float*)d_ws;        // [N,64]  h (initial embedding, fp32)
  float* buf_x = buf_h + NH;          // [N,64]  fp32 intermediate (feeds prop)
  float* buf_p = buf_x + NH;          // [N,128] concat(x1|x2) atomic accumulator
  // bf16 buffers (consumed only as WMMA A-operands)
  bf16_t* buf_hb = (bf16_t*)(buf_p + 2 * NH);  // [N,64] bf16 shadow of h
  bf16_t* bufb_x = buf_hb + NH;                // [N,64]
  bf16_t* bufb_t = bufb_x + NH;                // [N,64]
  // bf16 weight staging
  bf16_t* wb_first = bufb_t + NH;
  bf16_t* wb_con1  = wb_first + 64 * 128;
  bf16_t* wb_con2  = wb_con1  + 64 * 128;
  bf16_t* wb_lin1  = wb_con2  + 64 * 128;
  bf16_t* wb_out   = wb_lin1  + 64 * 64;
  bf16_t* wb_ih    = wb_out   + 16 * 64;
  bf16_t* wb_hh    = wb_ih    + 192 * 64;

  const int gemmBlocks = (N + 63) / 64;        // 4 waves/block, 16 rows/wave
  const int gruBlocks  = (N + 31) / 32;        // 2 waves/block
  const long long propThreads = (long long)E * 16;
  const int propBlocks = (int)((propThreads + 255) / 256);
  const size_t p4 = (size_t)N * 128 / 4;
  const int zeroBlocks = (int)((p4 + 255) / 256);

  // Pre-convert weights to bf16 (tiny, once per launch)
  auto cvt = [&](const float* s, bf16_t* dth, int n) {
    f32_to_bf16_kernel<<<(n + 255) / 256, 256, 0, stream>>>(s, dth, n);
  };
  cvt(W_first, wb_first, 64 * 128);
  cvt(W_con1,  wb_con1,  64 * 128);
  cvt(W_con2,  wb_con2,  64 * 128);
  cvt(W_lin1,  wb_lin1,  64 * 64);
  cvt(W_out,   wb_out,   16 * 64);
  cvt(W_ih,    wb_ih,    192 * 64);
  cvt(W_hh,    wb_hh,    192 * 64);

  // Layer 0: h = x @ W_first^T + b_first  (write fp32 + bf16 shadow)
  gemm_bias_kernel<128, 64, float><<<gemmBlocks, 128, 0, stream>>>(
      x_in, wb_first, b_first, buf_h, buf_hb, N);

  // Layer 1 propagation (fwd -> cols 0..63, rev -> cols 64..127)
  zero4_kernel<<<zeroBlocks, 256, 0, stream>>>((float4*)buf_p, p4);
  prop_kernel<<<propBlocks, 256, 0, stream>>>(buf_h, ei,   ew,   buf_p, 0,  E);
  prop_kernel<<<propBlocks, 256, 0, stream>>>(buf_h, ei_r, ew_r, buf_p, 64, E);
  gemm_bias_kernel<128, 64, float><<<gemmBlocks, 128, 0, stream>>>(
      buf_p, wb_con1, b_con1, nullptr, bufb_x, N);
  gru_kernel<<<gruBlocks, 64, 0, stream>>>(bufb_x, buf_hb, buf_h, wb_ih, wb_hh,
                                           b_ih, b_hh, bufb_t, N);
  gemm_bias_kernel<64, 64, bf16_t><<<gemmBlocks, 128, 0, stream>>>(
      bufb_t, wb_lin1, b_lin1, buf_x, nullptr, N);   // fp32: feeds propagation

  // Layer 2 propagation
  zero4_kernel<<<zeroBlocks, 256, 0, stream>>>((float4*)buf_p, p4);
  prop_kernel<<<propBlocks, 256, 0, stream>>>(buf_x, ei,   ew,   buf_p, 0,  E);
  prop_kernel<<<propBlocks, 256, 0, stream>>>(buf_x, ei_r, ew_r, buf_p, 64, E);
  gemm_bias_kernel<128, 64, float><<<gemmBlocks, 128, 0, stream>>>(
      buf_p, wb_con2, b_con2, nullptr, bufb_t, N);
  gru_kernel<<<gruBlocks, 64, 0, stream>>>(bufb_t, buf_hb, buf_h, wb_ih, wb_hh,
                                           b_ih, b_hh, bufb_x, N);

  // Output head + log_softmax (C = 16)
  gemm_bias_kernel<64, 16, bf16_t><<<gemmBlocks, 128, 0, stream>>>(
      bufb_x, wb_out, b_out, (float*)d_out, nullptr, N);
  logsoftmax16_kernel<<<(N + 255) / 256, 256, 0, stream>>>((float*)d_out, N);
}
